// GIN_72541997630002
// MI455X (gfx1250) — matrix operand order repaired and checked by reading, not compiled, not run on hardware
//
#include <hip/hip_runtime.h>
#include <hip/hip_bf16.h>

// ---------------------------------------------------------------------------
// GIN forward pass for MI455X (gfx1250, wave32, WMMA).
//  - Dense GEMMs: v_wmma_f32_16x16x32_f16, A tile staged once per block into
//    LDS (8 waves share it), weights pre-transposed to f16 (L2-resident).
//  - GEMM1 -> GEMM2 intermediate kept in f16 (halves HBM traffic).
//  - f16 A-tile staging uses global_load_async_to_lds_b128 when available.
//  - Sparse aggregation / pooling: float4 gathers + global_atomic_add_f32.
// ---------------------------------------------------------------------------

typedef __attribute__((ext_vector_type(16))) _Float16 v16h;
typedef __attribute__((ext_vector_type(8)))  _Float16 v8h;
typedef __attribute__((ext_vector_type(8)))  float    v8f;
typedef __attribute__((ext_vector_type(4)))  int      v4i;

#define HID 128
#define NLAYERS 3
#define BN_EPS 1e-5f

#if __has_builtin(__builtin_amdgcn_global_load_async_to_lds_b128) && \
    __has_builtin(__builtin_amdgcn_s_wait_asynccnt)
#define HAVE_ASYNC_LDS 1
typedef __attribute__((address_space(1))) v4i* gbl_v4i_p;  // global (__device__)
typedef __attribute__((address_space(3))) v4i* lds_v4i_p;  // LDS (__shared__)
#else
#define HAVE_ASYNC_LDS 0
#endif

static __device__ __forceinline__ size_t gtid() {
  return (size_t)blockIdx.x * blockDim.x + threadIdx.x;
}

// ---------------- small helpers ----------------

__global__ void zero_kernel(float* p, size_t n) {
  size_t i = gtid();
  if (i < n) p[i] = 0.0f;
}

// x[i,:] = z_table[z[i],:]   (float4 per thread)
__global__ void embed_kernel(const int* __restrict__ z,
                             const float* __restrict__ table,
                             float* __restrict__ x, int N) {
  size_t idx = gtid();
  if (idx >= (size_t)N * (HID / 4)) return;
  int node = (int)(idx >> 5);
  int f4   = (int)(idx & 31);
  const float4 v = *(const float4*)(table + (size_t)z[node] * HID + f4 * 4);
  *(float4*)(x + (size_t)node * HID + f4 * 4) = v;
}

__global__ void copy_f4(const float* __restrict__ src, float* __restrict__ dst,
                        size_t n4) {
  size_t i = gtid();
  if (i < n4) *(float4*)(dst + i * 4) = *(const float4*)(src + i * 4);
}

// agg[dst[e],:] += x[src[e],:]
__global__ void edge_scatter(const int* __restrict__ src,
                             const int* __restrict__ dst,
                             const float* __restrict__ x,
                             float* __restrict__ agg, int E) {
  size_t idx = gtid();
  if (idx >= (size_t)E * (HID / 4)) return;
  int e  = (int)(idx >> 5);
  int f4 = (int)(idx & 31);
  int s = src[e], d = dst[e];
  float4 v = *(const float4*)(x + (size_t)s * HID + f4 * 4);
  float* p = agg + (size_t)d * HID + f4 * 4;
  atomicAdd(p + 0, v.x);
  atomicAdd(p + 1, v.y);
  atomicAdd(p + 2, v.z);
  atomicAdd(p + 3, v.w);
}

// Convert + transpose weights to half, column-major (col*K + k) so WMMA
// B-fragments are two contiguous 16B loads per lane.
__global__ void prep_weights(const float* __restrict__ W1,
                             const float* __restrict__ W2,
                             const float* __restrict__ lin1,
                             _Float16* __restrict__ W1T,
                             _Float16* __restrict__ W2T,
                             _Float16* __restrict__ lin1T) {
  int idx = (int)gtid();
  const int convN = NLAYERS * HID * HID;
  if (idx < convN) {
    int l   = idx / (HID * HID);
    int rem = idx % (HID * HID);
    int n = rem / HID;   // output col
    int k = rem % HID;   // input k
    W1T[(size_t)l * HID * HID + (size_t)n * HID + k] =
        (_Float16)W1[(size_t)l * HID * HID + (size_t)k * HID + n];
    W2T[(size_t)l * HID * HID + (size_t)n * HID + k] =
        (_Float16)W2[(size_t)l * HID * HID + (size_t)k * HID + n];
  } else {
    int j = idx - convN;
    const int K1 = NLAYERS * HID; // 384
    if (j < HID * K1) {
      int n = j / K1;
      int k = j % K1;
      lin1T[(size_t)n * K1 + k] = (_Float16)lin1[(size_t)k * HID + n];
    }
  }
}

// ---------------- WMMA GEMM: C[rows,128] = act(A[rows,K] @ W[K,128] + b) -----
// WT: half, column-major WT[col*K + k].  TIN in {float, _Float16},
// TOUT in {float, _Float16}.  A tile (16 x K) staged in LDS once per block,
// padded row stride (K+8 halves) to dodge bank conflicts; 8 waves consume it.
// Grid: rows/16 blocks x 256 threads; wave w owns column tile w.
template <typename TIN, typename TOUT>
__global__ __launch_bounds__(256)
void gemm128_wmma(const TIN* __restrict__ A, int lda,
                  const _Float16* __restrict__ WT,
                  const float* __restrict__ bias,
                  TOUT* __restrict__ C, int ldc,
                  int K, int do_relu) {
  extern __shared__ char smem_raw[];
  _Float16* sA = (_Float16*)smem_raw;          // [16][K+8]
  const int LDS_STRIDE = K + 8;

  const int tid  = threadIdx.x;
  const int row0 = blockIdx.x << 4;

  // ---- cooperative stage of A tile: 16*K elements, 8 per thread per pass ----
  constexpr bool in_half = __is_same(TIN, _Float16);
  const int total = 16 * K;
  for (int i = tid * 8; i < total; i += 256 * 8) {
    const int r = i / K;
    const int k = i % K;                       // multiple of 8
    const TIN* gsrc = A + (size_t)(row0 + r) * lda + k;
    _Float16*  ldst = sA + r * LDS_STRIDE + k; // 16B aligned
    if constexpr (in_half) {
#if HAVE_ASYNC_LDS
      // async DMA of 16B chunk into LDS (tracked by ASYNCcnt)
      __builtin_amdgcn_global_load_async_to_lds_b128(
          (gbl_v4i_p)gsrc, (lds_v4i_p)ldst, 0, 0);
#else
      *(v8h*)ldst = *(const v8h*)gsrc;
#endif
    } else {
      __builtin_prefetch(gsrc + 2 * lda, 0, 0);   // global_prefetch_b8
      const float4 f0 = *(const float4*)(gsrc);
      const float4 f1 = *(const float4*)(gsrc + 4);
      v8h h;
      h[0] = (_Float16)f0.x; h[1] = (_Float16)f0.y;
      h[2] = (_Float16)f0.z; h[3] = (_Float16)f0.w;
      h[4] = (_Float16)f1.x; h[5] = (_Float16)f1.y;
      h[6] = (_Float16)f1.z; h[7] = (_Float16)f1.w;
      *(v8h*)ldst = h;
    }
  }
#if HAVE_ASYNC_LDS
  if constexpr (in_half) __builtin_amdgcn_s_wait_asynccnt(0);
#endif
  __syncthreads();

  // ---- WMMA main loop ----
  const int wave = tid >> 5;
  const int lane = tid & 31;
  const int l15  = lane & 15;
  const int hi   = lane >> 4;                  // 0: lanes 0-15, 1: 16-31
  const int col0 = wave << 4;
  const int bcol = col0 + l15;

  const _Float16* Wcol = WT + (size_t)bcol * K;
  const _Float16* Srow = sA + l15 * LDS_STRIDE;

  v8f c = {};
  for (int kk = 0; kk < K; kk += 32) {
    // A fragment: halves 0-7 = K=kbase..+7, 8-15 = K=kbase+16..+23
    const int kbase = kk + hi * 8;
    const v8h alo = *(const v8h*)(Srow + kbase);        // ds_load_b128
    const v8h ahi = *(const v8h*)(Srow + kbase + 16);   // ds_load_b128
    const v16h a = __builtin_shufflevector(alo, ahi, 0, 1, 2, 3, 4, 5, 6, 7,
                                           8, 9, 10, 11, 12, 13, 14, 15);
    // B fragment: halves 0-7 = K=kbase..+7, 8-15 = K=kbase+16..+23
    const v8h blo = *(const v8h*)(Wcol + kbase);
    const v8h bhi = *(const v8h*)(Wcol + kbase + 16);
    const v16h b = __builtin_shufflevector(blo, bhi, 0, 1, 2, 3, 4, 5, 6, 7,
                                           8, 9, 10, 11, 12, 13, 14, 15);
    c = __builtin_amdgcn_wmma_f32_16x16x32_f16(
        /*neg_a=*/false, a, /*neg_b=*/false, b,
        /*c_mod=*/(short)0, c, /*reuse_a=*/false, /*reuse_b=*/false);
  }

  // Epilogue: VGPR r -> row row0 + r + hi*8, col = bcol
  const float bv = bias[bcol];
  const int rbase = row0 + hi * 8;
#pragma unroll
  for (int r = 0; r < 8; ++r) {
    float v = c[r] + bv;
    if (do_relu) v = fmaxf(v, 0.0f);
    C[(size_t)(rbase + r) * ldc + bcol] = (TOUT)v;
  }
}

// ---------------- BatchNorm (training-mode batch statistics) ----------------

__global__ __launch_bounds__(256)
void bn_stats(const float* __restrict__ h, float* __restrict__ stats, int N) {
  __shared__ float ssum[HID];
  __shared__ float ssq[HID];
  if (threadIdx.x < HID) { ssum[threadIdx.x] = 0.0f; ssq[threadIdx.x] = 0.0f; }
  __syncthreads();
  const size_t total = (size_t)N * (HID / 4);
  for (size_t idx = gtid(); idx < total; idx += (size_t)gridDim.x * blockDim.x) {
    const int f = (int)(idx & 31) * 4;
    const float4 v = *(const float4*)(h + idx * 4);
    atomicAdd(&ssum[f + 0], v.x); atomicAdd(&ssq[f + 0], v.x * v.x);
    atomicAdd(&ssum[f + 1], v.y); atomicAdd(&ssq[f + 1], v.y * v.y);
    atomicAdd(&ssum[f + 2], v.z); atomicAdd(&ssq[f + 2], v.z * v.z);
    atomicAdd(&ssum[f + 3], v.w); atomicAdd(&ssq[f + 3], v.w * v.w);
  }
  __syncthreads();
  if (threadIdx.x < HID) {
    atomicAdd(&stats[threadIdx.x],       ssum[threadIdx.x]);
    atomicAdd(&stats[HID + threadIdx.x], ssq[threadIdx.x]);
  }
}

__global__ void bn_apply(float* __restrict__ h, const float* __restrict__ stats,
                         const float* __restrict__ gamma,
                         const float* __restrict__ beta, int N) {
  size_t idx = gtid();
  if (idx >= (size_t)N * HID) return;
  const int f = (int)(idx & (HID - 1));
  const float invN = 1.0f / (float)N;
  const float mu   = stats[f] * invN;
  const float var  = stats[HID + f] * invN - mu * mu;
  const float rstd = rsqrtf(var + BN_EPS);
  h[idx] = gamma[f] * (h[idx] - mu) * rstd + beta[f];
}

// ---------------- mean pooling over graphs ----------------

__global__ void pool_count(const int* __restrict__ batch, float* __restrict__ cnt,
                           int N) {
  size_t i = gtid();
  if (i < (size_t)N) atomicAdd(&cnt[batch[i]], 1.0f);
}

__global__ void pool_scatter(const float* __restrict__ X0,
                             const float* __restrict__ X1,
                             const float* __restrict__ X2,
                             const int* __restrict__ batch,
                             float* __restrict__ sums, int N) {
  size_t idx = gtid();
  if (idx >= (size_t)N * 96) return;            // 96 float4s = 384 feats
  const int node = (int)(idx / 96);
  const int c    = (int)(idx % 96);
  const int l    = c >> 5;
  const int f4   = c & 31;
  const float* X = (l == 0) ? X0 : (l == 1) ? X1 : X2;
  const float4 v = *(const float4*)(X + (size_t)node * HID + f4 * 4);
  float* p = sums + (size_t)batch[node] * (NLAYERS * HID) + l * HID + f4 * 4;
  atomicAdd(p + 0, v.x);
  atomicAdd(p + 1, v.y);
  atomicAdd(p + 2, v.z);
  atomicAdd(p + 3, v.w);
}

__global__ void pool_div(float* __restrict__ sums, const float* __restrict__ cnt,
                         int G) {
  size_t idx = gtid();
  if (idx >= (size_t)G * (NLAYERS * HID)) return;
  const int g = (int)(idx / (NLAYERS * HID));
  sums[idx] /= fmaxf(cnt[g], 1.0f);
}

// out[g] = dot(hidden[g,:], w) + b   — one wave per graph, shuffle reduce
__global__ void mlp2_kernel(const float* __restrict__ hidden,
                            const float* __restrict__ w,
                            const float* __restrict__ b,
                            float* __restrict__ out, int G) {
  const int wid  = (int)(gtid() >> 5);
  const int lane = threadIdx.x & 31;
  if (wid >= G) return;
  const float4 h4 = *(const float4*)(hidden + (size_t)wid * HID + lane * 4);
  const float4 w4 = *(const float4*)(w + lane * 4);
  float s = h4.x * w4.x + h4.y * w4.y + h4.z * w4.z + h4.w * w4.w;
  for (int off = 16; off > 0; off >>= 1) s += __shfl_down(s, off, 32);
  if (lane == 0) out[wid] = s + b[0];
}

// ---------------------------------------------------------------------------

static inline size_t align256(size_t x) { return (x + 255) & ~(size_t)255; }

extern "C" void kernel_launch(void* const* d_in, const int* in_sizes, int n_in,
                              void* d_out, int out_size, void* d_ws, size_t ws_size,
                              hipStream_t stream) {
  (void)n_in; (void)ws_size;
  const int*   z       = (const int*)d_in[0];
  const int*   ei      = (const int*)d_in[1];
  const int*   batch   = (const int*)d_in[2];
  const float* z_table = (const float*)d_in[3];
  const float* W1      = (const float*)d_in[4];
  const float* b1      = (const float*)d_in[5];
  const float* W2      = (const float*)d_in[6];
  const float* b2      = (const float*)d_in[7];
  const float* gamma   = (const float*)d_in[8];
  const float* beta    = (const float*)d_in[9];
  const float* lin1W   = (const float*)d_in[10];
  const float* lin1b   = (const float*)d_in[11];
  const float* lin2W   = (const float*)d_in[12];
  const float* lin2b   = (const float*)d_in[13];
  float* out = (float*)d_out;

  const int N = in_sizes[0];
  const int E = in_sizes[1] / 2;
  const int G = out_size;                 // [G,1] output
  const int* src = ei;
  const int* dst = ei + E;

  const size_t NH = (size_t)N * HID;

  // ---- workspace layout (all 256B aligned) ----
  char* base = (char*)d_ws;
  size_t off = 0;
  float* xemb  = (float*)(base + off); off = align256(off + NH * 4);  // also f16 temp
  float* agg   = (float*)(base + off); off = align256(off + NH * 4);
  float* X0    = (float*)(base + off); off = align256(off + NH * 4);
  float* X1    = (float*)(base + off); off = align256(off + NH * 4);
  float* X2    = (float*)(base + off); off = align256(off + NH * 4);
  float* stats = (float*)(base + off); off = align256(off + 2 * HID * 4);
  float* psum  = (float*)(base + off); off = align256(off + (size_t)G * NLAYERS * HID * 4);
  float* pcnt  = (float*)(base + off); off = align256(off + (size_t)G * 4);
  float* hid   = (float*)(base + off); off = align256(off + (size_t)G * HID * 4);
  _Float16* W1T   = (_Float16*)(base + off); off = align256(off + (size_t)NLAYERS * HID * HID * 2);
  _Float16* W2T   = (_Float16*)(base + off); off = align256(off + (size_t)NLAYERS * HID * HID * 2);
  _Float16* lin1T = (_Float16*)(base + off); off = align256(off + (size_t)HID * NLAYERS * HID * 2);

  _Float16* htmp16 = (_Float16*)xemb;   // GEMM1->GEMM2 intermediate (aliases xemb)

  const int BLK = 256;
  const int LDS128 = 16 * (HID + 8) * 2;           // 4352 B
  const int LDS384 = 16 * (NLAYERS * HID + 8) * 2; // 12544 B
  float* Xl[NLAYERS] = {X0, X1, X2};

  // ---- weight prep (f32 -> half, transposed) ----
  {
    const int tot = NLAYERS * HID * HID + HID * NLAYERS * HID;
    prep_weights<<<(tot + BLK - 1) / BLK, BLK, 0, stream>>>(W1, W2, lin1W, W1T, W2T, lin1T);
  }

  // ---- embedding lookup ----
  {
    const size_t t = (size_t)N * (HID / 4);
    embed_kernel<<<(unsigned)((t + BLK - 1) / BLK), BLK, 0, stream>>>(z, z_table, xemb, N);
  }

  // ---- GIN layers ----
  const float* x_in = xemb;
  for (int l = 0; l < NLAYERS; ++l) {
    // agg = x + segment_sum(x[src], dst)
    {
      const size_t t = NH / 4;
      copy_f4<<<(unsigned)((t + BLK - 1) / BLK), BLK, 0, stream>>>(x_in, agg, t);
      const size_t te = (size_t)E * (HID / 4);
      edge_scatter<<<(unsigned)((te + BLK - 1) / BLK), BLK, 0, stream>>>(src, dst, x_in, agg, E);
    }
    // h = relu(agg @ W1 + b1)   (WMMA, f32 in -> f16 out; xemb free after scatter)
    gemm128_wmma<float, _Float16><<<N / 16, BLK, LDS128, stream>>>(
        agg, HID, W1T + (size_t)l * HID * HID, b1 + l * HID, htmp16, HID, HID, 1);
    // h = relu(h @ W2 + b2)     (WMMA, f16 in via async-LDS -> f32 out)
    gemm128_wmma<_Float16, float><<<N / 16, BLK, LDS128, stream>>>(
        htmp16, HID, W2T + (size_t)l * HID * HID, b2 + l * HID, Xl[l], HID, HID, 1);
    // BatchNorm over nodes, in place on X[l]
    zero_kernel<<<1, BLK, 0, stream>>>(stats, 2 * HID);
    bn_stats<<<1024, BLK, 0, stream>>>(Xl[l], stats, N);
    {
      const size_t t = NH;
      bn_apply<<<(unsigned)((t + BLK - 1) / BLK), BLK, 0, stream>>>(
          Xl[l], stats, gamma + l * HID, beta + l * HID, N);
    }
    x_in = Xl[l];
  }

  // ---- global mean pool over [N, 384] ----
  {
    const size_t nz = (size_t)G * NLAYERS * HID + (size_t)G;  // psum + pcnt contiguous
    zero_kernel<<<(unsigned)((nz + BLK - 1) / BLK), BLK, 0, stream>>>(psum, nz);
    pool_count<<<(N + BLK - 1) / BLK, BLK, 0, stream>>>(batch, pcnt, N);
    const size_t ts = (size_t)N * 96;
    pool_scatter<<<(unsigned)((ts + BLK - 1) / BLK), BLK, 0, stream>>>(X0, X1, X2, batch, psum, N);
    const size_t td = (size_t)G * NLAYERS * HID;
    pool_div<<<(unsigned)((td + BLK - 1) / BLK), BLK, 0, stream>>>(psum, pcnt, G);
  }

  // ---- hidden = relu(pooled @ lin1_W + lin1_b)   (WMMA, K=384) ----
  gemm128_wmma<float, float><<<G / 16, BLK, LDS384, stream>>>(
      psum, NLAYERS * HID, lin1T, lin1b, hid, HID, NLAYERS * HID, 1);

  // ---- out = hidden @ lin2_W + lin2_b ----
  mlp2_kernel<<<(G * 32 + BLK - 1) / BLK, BLK, 0, stream>>>(hid, lin2W, lin2b, out, G);
}
